// KolmogorovLayer_9723805958305
// MI455X (gfx1250) — matrix operand order repaired and checked by reading, not compiled, not run on hardware
//
#include <hip/hip_runtime.h>

// ---------------------------------------------------------------------------
// KolmogorovLayer fused kernel for MI455X (gfx1250, wave32, WMMA).
//
// Algebraic fusion:  psi@Wg1 == h @ (W2[d]@Wg1_d stacked)  -> single GEMM
//   hg = tanh( h[B,4096] @ M[4096,64] + c ),  out = hg @ Wg2 + bg2
// with h generated on the fly in registers: h[b,k] = tanh(x[b,k>>6]*W1f[k]+b1f[k]).
//
// Main GEMM runs on v_wmma_f32_16x16x32_f16 (fp32 accumulate). f16 chosen over
// bf16: identical WMMA rate, 3 more mantissa bits, and all operands are
// bounded (tanh outputs in (-1,1), M entries tiny sums of uniforms).
// tanh uses native v_tanh_f32 when the builtin exists (1 trans op), else the
// exp2+rcp fallback (2 trans ops).
// ---------------------------------------------------------------------------

typedef __attribute__((ext_vector_type(16))) _Float16 v16h;
typedef __attribute__((ext_vector_type(8)))  float    v8f;

#define BATCH 32768
#define DIMD  64
#define DIMI  64
#define KDIM  (DIMD * DIMI)   // 4096
#define KSTEPS (KDIM / 32)    // 128 k-steps of 32

__device__ __forceinline__ float tanh_fast(float x) {
#if __has_builtin(__builtin_amdgcn_tanhf)
    return __builtin_amdgcn_tanhf(x);          // v_tanh_f32 (CDNA5 trans op)
#else
    // tanh(x) = 1 - 2/(exp(2x)+1); exp(2x) = exp2(x * 2*log2(e))
    float e = __builtin_amdgcn_exp2f(x * 2.8853900817779268f);
    return 1.0f - 2.0f * __builtin_amdgcn_rcpf(e + 1.0f);
#endif
}

// ---------------------------------------------------------------------------
// Prep 1: M[d] = W2[d] @ Wg1_d, written as f16 directly in WMMA B-fragment
// order. B K order used here (contiguous, not AMD's table order): VGPR v holds K=2v,2v+1;
// lanes 0-15 -> K=0..15 (N=lane), lanes 16-31 -> K=16..31 (N=lane-16).
// Fragment element e (f16 #e of the contiguous 32B chunk) => K = half*16 + e.
// Mpack index: (((kstep*4 + ntile)*32 + lane) * 16 + e)
// ---------------------------------------------------------------------------
__global__ __launch_bounds__(256) void prep_M(const float* __restrict__ W2,
                                              const float* __restrict__ Wg1,
                                              _Float16* __restrict__ Mpack) {
    __shared__ float lW2[DIMI * DIMI];   // W2[d]  : [i][o]
    __shared__ float lWg[DIMI * DIMI];   // Wg1_d  : [o][j]
    const int d   = blockIdx.x;
    const int tid = threadIdx.x;
    #pragma unroll
    for (int r = 0; r < 16; ++r) {
        int idx = tid + r * 256;                       // [0,4096)
        lW2[idx] = W2[d * 4096 + idx];
        lWg[idx] = Wg1[(d * 64 + (idx >> 6)) * 64 + (idx & 63)];
    }
    __syncthreads();
    #pragma unroll
    for (int r = 0; r < 16; ++r) {
        int oidx = tid + r * 256;
        int i = oidx >> 6, j = oidx & 63;
        float s = 0.0f;
        #pragma unroll 8
        for (int o = 0; o < 64; ++o)
            s = fmaf(lW2[i * 64 + o], lWg[o * 64 + j], s);
        int k    = d * 64 + i;          // global K row of M
        int ks   = k >> 5;              // k-step
        int Kl   = k & 31;              // K within 32-wide step
        int hlf  = Kl >> 4;             // which half-lane group
        int e    = Kl & 15;             // element within lane chunk
        int lane = hlf * 16 + (j & 15);
        int t    = j >> 4;              // N tile
        Mpack[((((ks * 4 + t) * 32) + lane) << 4) + e] = (_Float16)s;
    }
}

// ---------------------------------------------------------------------------
// Prep 2: re-pack W1/b1 (flattened [4096]) into A-fragment K order.
// A-fragment K order matching Mpack: element e of half => K = half*16 + e
// stored as W1p[(kstep*2 + half)*16 + e].
// ---------------------------------------------------------------------------
__global__ void pack_w1b1(const float* __restrict__ W1, const float* __restrict__ b1,
                          float* __restrict__ W1p, float* __restrict__ b1p) {
    int idx = blockIdx.x * 256 + threadIdx.x;          // [0,4096)
    int e = idx & 15, hlf = (idx >> 4) & 1, s = idx >> 5;
    int Kl = hlf * 16 + e;
    int k = s * 32 + Kl;
    W1p[idx] = W1[k];
    b1p[idx] = b1[k];
}

// Prep 3: fused bias vector c[j] = bg1[j] + sum_q b2_flat[q] * Wg1[q][j]
__global__ void prep_c(const float* __restrict__ b2, const float* __restrict__ Wg1,
                       const float* __restrict__ bg1, float* __restrict__ cvec) {
    int j = threadIdx.x;                               // 64 threads
    float s = bg1[j];
    for (int q = 0; q < KDIM; ++q)
        s = fmaf(b2[q], Wg1[q * 64 + j], s);
    cvec[j] = s;
}

// ---------------------------------------------------------------------------
// Main fused kernel: 256 threads = 8 waves; each wave owns 16 rows x 64 cols
// (4 accumulator tiles -> each A fragment feeds 4 WMMAs). WG covers 128 rows.
// ---------------------------------------------------------------------------
__global__ __launch_bounds__(256, 1) void kan_fused(
    const float*    __restrict__ x,
    const float*    __restrict__ W1p,
    const float*    __restrict__ b1p,
    const _Float16* __restrict__ Mpack,
    const float*    __restrict__ cvec,
    const float*    __restrict__ Wg2,
    const float*    __restrict__ bg2,
    float*          __restrict__ out) {

    const int lane = threadIdx.x & 31;
    const int wave = threadIdx.x >> 5;
    const int hlf  = lane >> 4;                 // half-lane group
    const int rbase = blockIdx.x * 128 + wave * 16;
    const int mrow  = rbase + (lane & 15);      // A-matrix row this lane feeds

    v8f acc[4];
    #pragma unroll
    for (int t = 0; t < 4; ++t)
        #pragma unroll
        for (int v = 0; v < 8; ++v) acc[t][v] = 0.0f;

    const v16h* __restrict__ Mfrag = (const v16h*)Mpack;

    for (int d = 0; d < DIMD; ++d) {
        // one x scalar per lane per feature; k-window [d*64, d*64+64)
        const float xv = x[mrow * DIMD + d];

        #pragma unroll
        for (int sub = 0; sub < 2; ++sub) {
            const int s = d * 2 + sub;          // k-step of 32

            const float4* w1c = (const float4*)(W1p + ((s * 2 + hlf) << 4));
            const float4* b1c = (const float4*)(b1p + ((s * 2 + hlf) << 4));

            // generate this lane's 16 A elements: h = tanh(x*w1 + b1) -> f16
            v16h af;
            #pragma unroll
            for (int q = 0; q < 4; ++q) {
                float4 w = w1c[q], b = b1c[q];
                af[q * 4 + 0] = (_Float16)tanh_fast(fmaf(xv, w.x, b.x));
                af[q * 4 + 1] = (_Float16)tanh_fast(fmaf(xv, w.y, b.y));
                af[q * 4 + 2] = (_Float16)tanh_fast(fmaf(xv, w.z, b.z));
                af[q * 4 + 3] = (_Float16)tanh_fast(fmaf(xv, w.w, b.w));
            }

            // prefetch next k-step's B fragments (global_prefetch)
            if (s + 1 < KSTEPS)
                __builtin_prefetch(&Mfrag[(s + 1) * 4 * 32 + lane], 0, 1);

            #pragma unroll
            for (int t = 0; t < 4; ++t) {
                v16h bf = Mfrag[(s * 4 + t) * 32 + lane];   // coalesced 32B/lane
                acc[t] = __builtin_amdgcn_wmma_f32_16x16x32_f16(
                    false, af, false, bf, (short)0, acc[t], false, false);
            }
        }
    }

    // Epilogue. C layout: N = (lane&15) + 16*t ; M = v + 8*hlf.
    // hg = tanh(acc + c); partial row-dot with Wg2; butterfly-reduce over N.
    float cn[4], g2[4];
    #pragma unroll
    for (int t = 0; t < 4; ++t) {
        int n = t * 16 + (lane & 15);
        cn[t] = cvec[n];
        g2[t] = Wg2[n];
    }
    float part[8];
    #pragma unroll
    for (int v = 0; v < 8; ++v) {
        float sum = 0.0f;
        #pragma unroll
        for (int t = 0; t < 4; ++t)
            sum = fmaf(tanh_fast(acc[t][v] + cn[t]), g2[t], sum);
        part[v] = sum;
    }
    // reduce across the 16 lanes of each half (xor of bits 0..3 stays in-group)
    #pragma unroll
    for (int off = 1; off < 16; off <<= 1)
        #pragma unroll
        for (int v = 0; v < 8; ++v)
            part[v] += __shfl_xor(part[v], off, 32);

    if ((lane & 15) == 0) {
        float bb = bg2[0];
        #pragma unroll
        for (int v = 0; v < 8; ++v)
            out[rbase + hlf * 8 + v] = part[v] + bb;
    }
}

// ---------------------------------------------------------------------------
// Workspace layout (all hipMalloc-aligned offsets):
//   [0, 512KB)          Mpack  f16[4096*64]
//   [512KB, 528KB)      W1p    f32[4096]
//   [528KB, 544KB)      b1p    f32[4096]
//   [544KB, +256B)      cvec   f32[64]
// ---------------------------------------------------------------------------
extern "C" void kernel_launch(void* const* d_in, const int* in_sizes, int n_in,
                              void* d_out, int out_size, void* d_ws, size_t ws_size,
                              hipStream_t stream) {
    const float* x   = (const float*)d_in[0];
    const float* W1  = (const float*)d_in[1];
    const float* b1  = (const float*)d_in[2];
    const float* W2  = (const float*)d_in[3];
    const float* b2  = (const float*)d_in[4];
    const float* Wg1 = (const float*)d_in[5];
    const float* bg1 = (const float*)d_in[6];
    const float* Wg2 = (const float*)d_in[7];
    const float* bg2 = (const float*)d_in[8];
    float* out = (float*)d_out;

    char* ws = (char*)d_ws;
    _Float16* Mpack = (_Float16*)(ws);
    float*    W1p   = (float*)(ws + (512 << 10));
    float*    b1p   = (float*)(ws + (512 << 10) + (16 << 10));
    float*    cvec  = (float*)(ws + (512 << 10) + (32 << 10));

    prep_M   <<<DIMD, 256, 0, stream>>>(W2, Wg1, Mpack);
    pack_w1b1<<<16,   256, 0, stream>>>(W1, b1, W1p, b1p);
    prep_c   <<<1,     64, 0, stream>>>(b2, Wg1, bg1, cvec);
    kan_fused<<<BATCH / 128, 256, 0, stream>>>(x, W1p, b1p, Mpack, cvec, Wg2, bg2, out);
}
